// PairNetLayer_20839181320727
// MI455X (gfx1250) — matrix operand order repaired and checked by reading, not compiled, not run on hardware
//
#include <hip/hip_runtime.h>
#include <hip/hip_bf16.h>

// ---------------------------------------------------------------------------
// PairNetLayer for MI455X (gfx1250), wave32 + bf16 WMMA (f32 accumulate).
//   - all weights pre-packed to bf16 in B-fragment order (2x b128 per frag)
//   - all activations bf16, PLANAR layout [s|vx|vy|vz] so A-fragments are
//     two contiguous 16B chunks per lane (2x ds_load_b128, no conversions)
//   - interleaved e3nn layout only at input read and final residual store
// ---------------------------------------------------------------------------

typedef __attribute__((ext_vector_type(16))) __bf16 v16bf;
typedef __attribute__((ext_vector_type(8)))  __bf16 v8bf;
typedef __attribute__((ext_vector_type(8)))  float  v8f;

#define INV_SQ3 0.5773502691896258f
#define INV_SQ6 0.4082482904638631f
#define CW0     0.4472135954999579f   // sqrt(1/5)
#define CW1     0.7745966692414834f   // sqrt(3/5)
#define INV_S32 0.17677669529663687f  // 1/sqrt(32)
#define INV_S8  0.35355339059327373f  // 1/sqrt(8)
#define LN2F    0.6931471805599453f

// Per-wave LDS slab (bytes):
//   Wf : f32  [16x320] @ 0      (20480)  fs->w staging, later o3 f32 staging
//   Xb : bf16 [4][16][64] @20480 ( 8192)  planar activation tile (also s0 [16x192])
//   Fb : bf16 [16x128] @ 28672  ( 4096)  f0 / g
//   Gb : bf16 [16x128] @ 32768  ( 4096)  g1
//   Db : bf16 [16x64]  @ 36864  ( 2048)  fs-mid / padded fe input [16x32]
#define SLAB_B 38912
#define WAVES  4

__device__ __forceinline__ float siluf(float x) { return x / (1.f + __expf(-x)); }
__device__ __forceinline__ float sspf(float x)  { return __logf(1.f + __expf(x)) - LN2F; }
__device__ __forceinline__ v8f v8zero() {
  v8f z = {0.f,0.f,0.f,0.f,0.f,0.f,0.f,0.f};
  return z;
}

// A fragment (16x32 bf16) from a K-contiguous bf16 tile: lane = M row,
// half-wave picks K sub-block; per ISA layout the 16 elements are two
// contiguous 8-element runs at kBase+8*half and kBase+8*half+16.
__device__ __forceinline__ v16bf load_fragA(const __bf16* T, int ldT, int kBase, int lane) {
  const int row = lane & 15, half = lane >> 4;
  union { v16bf v; v8bf h[2]; } f;
  const __bf16* p = T + row * ldT + kBase + 8 * half;
  f.h[0] = *(const v8bf*)(p);
  f.h[1] = *(const v8bf*)(p + 16);
  return f.v;
}

// B fragment from pre-packed weights: 32 contiguous bytes per lane.
__device__ __forceinline__ v16bf load_fragB(const __bf16* Bp, int frag, int lane) {
  return ((const v16bf*)Bp)[(size_t)frag * 32 + lane];
}

// D += A(16x32K) * B(32Kx16) over KB k-steps; B packed as [nt][kb][lane][16].
__device__ __forceinline__ v8f gemm_tile(const __bf16* A, int ldA, const __bf16* Bp,
                                         int nt, int KB, int lane, v8f acc) {
#pragma unroll
  for (int kb = 0; kb < KB; ++kb) {
    v16bf a = load_fragA(A, ldA, kb * 32, lane);
    v16bf b = load_fragB(Bp, nt * KB + kb, lane);
    acc = __builtin_amdgcn_wmma_f32_16x16x32_bf16(false, a, false, b, (short)0, acc,
                                                  false, false);
  }
  return acc;
}

// o3.Linear on planar bf16 tile -> planar bf16 global rows (node stage).
__device__ __forceinline__ void o3_planar_out(const __bf16* Xb, const __bf16* W0p,
                                              const float* b0, const __bf16* W1p,
                                              __bf16* dst, int lane) {
  const int half = lane >> 4, nc = lane & 15;
  for (int nt = 0; nt < 4; ++nt) {                       // 0e plane
    v8f acc = gemm_tile(Xb, 64, W0p, nt, 2, lane, v8zero());
    float bias = b0[nt * 16 + nc];
#pragma unroll
    for (int r = 0; r < 8; ++r)
      dst[(size_t)(r + 8 * half) * 256 + nt * 16 + nc] = (__bf16)(acc[r] * 0.125f + bias);
  }
  for (int c = 0; c < 3; ++c)                            // 1e planes
    for (int nt = 0; nt < 4; ++nt) {
      v8f acc = gemm_tile(Xb + (1 + c) * 1024, 64, W1p, nt, 2, lane, v8zero());
#pragma unroll
      for (int r = 0; r < 8; ++r)
        dst[(size_t)(r + 8 * half) * 256 + (1 + c) * 64 + nt * 16 + nc] =
            (__bf16)(acc[r] * 0.125f);
    }
}

// o3.Linear on planar bf16 tile -> f32 LDS tile in reference interleaved layout.
__device__ __forceinline__ void o3_interleaved_out(const __bf16* Xb, const __bf16* W0p,
                                                   const float* b0, const __bf16* W1p,
                                                   float* dst, int lane) {
  const int half = lane >> 4, nc = lane & 15;
  for (int nt = 0; nt < 4; ++nt) {
    v8f acc = gemm_tile(Xb, 64, W0p, nt, 2, lane, v8zero());
    float bias = b0[nt * 16 + nc];
#pragma unroll
    for (int r = 0; r < 8; ++r)
      dst[(r + 8 * half) * 256 + nt * 16 + nc] = acc[r] * 0.125f + bias;
  }
  for (int c = 0; c < 3; ++c)
    for (int nt = 0; nt < 4; ++nt) {
      v8f acc = gemm_tile(Xb + (1 + c) * 1024, 64, W1p, nt, 2, lane, v8zero());
#pragma unroll
      for (int r = 0; r < 8; ++r)
        dst[(r + 8 * half) * 256 + 64 + (nt * 16 + nc) * 3 + c] = acc[r] * 0.125f;
    }
}

// NormGate on a planar bf16 tile, in place.
__device__ __forceinline__ void norm_gate_bf(__bf16* Xb, __bf16* Fb, __bf16* Gb,
                                             const __bf16* W1p, const float* b1,
                                             const __bf16* W2p, const float* b2,
                                             int lane) {
  for (int idx = lane; idx < 2048; idx += 32) {          // f0 = [s, |v_u|]
    int row = idx >> 7, j = idx & 127;
    float val;
    if (j < 64) {
      val = (float)Xb[row * 64 + j];
    } else {
      int u = j - 64;
      float a = (float)Xb[1024 + row * 64 + u];
      float b = (float)Xb[2048 + row * 64 + u];
      float c = (float)Xb[3072 + row * 64 + u];
      val = sqrtf(a * a + b * b + c * c);
    }
    Fb[idx] = (__bf16)val;
  }
  __syncthreads();
  const int half = lane >> 4, nc = lane & 15;
  for (int nt = 0; nt < 8; ++nt) {                       // g1 = silu(f0@W1+b1)
    v8f acc = gemm_tile(Fb, 128, W1p, nt, 4, lane, v8zero());
    float bias = b1[nt * 16 + nc];
#pragma unroll
    for (int r = 0; r < 8; ++r)
      Gb[(r + 8 * half) * 128 + nt * 16 + nc] = (__bf16)siluf(acc[r] + bias);
  }
  __syncthreads();
  for (int nt = 0; nt < 8; ++nt) {                       // g = g1@W2+b2 (into Fb)
    v8f acc = gemm_tile(Gb, 128, W2p, nt, 4, lane, v8zero());
    float bias = b2[nt * 16 + nc];
#pragma unroll
    for (int r = 0; r < 8; ++r)
      Fb[(r + 8 * half) * 128 + nt * 16 + nc] = (__bf16)(acc[r] + bias);
  }
  __syncthreads();
  for (int idx = lane; idx < 1024; idx += 32) {          // gate
    int row = idx >> 6, u = idx & 63;
    float g = (float)Fb[row * 128 + 64 + u];
    Xb[row * 64 + u] = Fb[row * 128 + u];
    Xb[1024 + row * 64 + u] = (__bf16)((float)Xb[1024 + row * 64 + u] * g);
    Xb[2048 + row * 64 + u] = (__bf16)((float)Xb[2048 + row * 64 + u] * g);
    Xb[3072 + row * 64 + u] = (__bf16)((float)Xb[3072 + row * 64 + u] * g);
  }
  __syncthreads();
}

// ------------------- weight pre-pack: f32 [K x N] -> bf16 B-fragments -------
__global__ void pack_weight(const float* __restrict__ W, __bf16* __restrict__ dst,
                            int ldW, int kMax, int KB, int NT) {
  int gid = blockIdx.x * 256 + threadIdx.x;
  if (gid >= KB * NT * 32) return;
  int lane = gid & 31, frag = gid >> 5;
  int kb = frag % KB, nt = frag / KB;
  int col = nt * 16 + (lane & 15), half = lane >> 4;
  __bf16* o = dst + (size_t)(nt * KB + kb) * 512 + lane * 16;
#pragma unroll
  for (int e = 0; e < 16; ++e) {
    int k = kb * 32 + 8 * half + (e & 7) + ((e >> 3) << 4);
    o[e] = (__bf16)((k < kMax) ? W[k * ldW + col] : 0.f);
  }
}

// ------------------------------- Stage 1: nodes ----------------------------
__global__ void __launch_bounds__(32 * WAVES) node_stage(
    const float* __restrict__ node_attr,
    const __bf16* Wi0p, const float* bi0, const __bf16* Wi1p,
    const __bf16* gpW1p, const float* gpb1, const __bf16* gpW2p, const float* gpb2,
    const __bf16* Wn0p, const float* bn0, const __bf16* Wn1p,
    __bf16* __restrict__ na0b, __bf16* __restrict__ hb,
    int Ntot, int nTiles) {
  extern __shared__ char smem[];
  const int wave = threadIdx.x >> 5, lane = threadIdx.x & 31;
  char* S = smem + wave * SLAB_B;
  __bf16* Xb = (__bf16*)(S + 20480);
  __bf16* Fb = (__bf16*)(S + 28672);
  __bf16* Gb = (__bf16*)(S + 32768);

  int tile = blockIdx.x * WAVES + wave;
  if (tile >= nTiles) tile = nTiles - 1;        // clamp: no barrier divergence
  int r0 = tile * 16;
  if (r0 + 16 > Ntot) r0 = Ntot - 16;

  // interleaved f32 -> planar bf16 tile
  for (int idx = lane; idx < 4096; idx += 32) {
    int p = idx >> 10, rem = idx & 1023, row = rem >> 6, u = rem & 63;
    int col = (p == 0) ? u : 64 + u * 3 + (p - 1);
    Xb[idx] = (__bf16)node_attr[(size_t)(r0 + row) * 256 + col];
  }
  __syncthreads();

  o3_planar_out(Xb, Wi0p, bi0, Wi1p, na0b + (size_t)r0 * 256, lane);      // na0
  __syncthreads();
  norm_gate_bf(Xb, Fb, Gb, gpW1p, gpb1, gpW2p, gpb2, lane);               // gate(x)
  o3_planar_out(Xb, Wn0p, bn0, Wn1p, hb + (size_t)r0 * 256, lane);        // h
}

// ------------------------------- Stage 2: edges ----------------------------
__global__ void __launch_bounds__(32 * WAVES) edge_stage(
    const float* __restrict__ edge_attr, const float* __restrict__ npa,
    const __bf16* __restrict__ na0b, const __bf16* __restrict__ hb,
    const float* feW1, const __bf16* feW2p,
    const __bf16* fsW1p, const float* fsb1, const __bf16* fsW2p, const float* fsb2,
    const __bf16* gW1p, const float* gb1, const __bf16* gW2p, const float* gb2,
    const __bf16* Wo0p, const float* bo0, const __bf16* Wo1p,
    const int* __restrict__ edge_index, float* __restrict__ out,
    int E, int eTiles) {
  extern __shared__ char smem[];
  const int wave = threadIdx.x >> 5, lane = threadIdx.x & 31;
  char* S = smem + wave * SLAB_B;
  float*  Wf = (float*)(S);             // [16x320] f32
  __bf16* Xb = (__bf16*)(S + 20480);    // planar pair tile; first acts as s0 [16x192]
  __bf16* Fb = (__bf16*)(S + 28672);
  __bf16* Gb = (__bf16*)(S + 32768);
  __bf16* Db = (__bf16*)(S + 36864);    // [16x64] / t [16x32]

  int tile = blockIdx.x * WAVES + wave;
  if (tile >= eTiles) tile = eTiles - 1;
  int e0 = tile * 16;
  if (e0 + 16 > E) e0 = E - 16;
  const int* dsts = edge_index;
  const int* srcs = edge_index + E;
  const int half = lane >> 4, nc = lane & 15;

  __builtin_prefetch(npa + (size_t)e0 * 256, 0, 1);

  // s0 = [s_dst, s_src, <v_dst,v_src>/3] -> bf16 [16x192] (L2-resident gathers)
  for (int idx = lane; idx < 3072; idx += 32) {
    int row = idx / 192, j = idx % 192;
    int e = e0 + row, d = dsts[e], s = srcs[e];
    float val;
    if (j < 64) {
      val = (float)na0b[(size_t)d * 256 + j];
    } else if (j < 128) {
      val = (float)na0b[(size_t)s * 256 + (j - 64)];
    } else {
      int u = j - 128;
      const __bf16* vd = na0b + (size_t)d * 256;
      const __bf16* vs = na0b + (size_t)s * 256;
      val = ((float)vd[64 + u] * (float)vs[64 + u] +
             (float)vd[128 + u] * (float)vs[128 + u] +
             (float)vd[192 + u] * (float)vs[192 + u]) * (1.f / 3.f);
    }
    Xb[idx] = (__bf16)val;
  }
  __syncthreads();

  // fs-mid = silu(s0 @ fsW1 + fsb1)  [16x192]@[192x64] -> Db
  for (int nt = 0; nt < 4; ++nt) {
    v8f acc = gemm_tile(Xb, 192, fsW1p, nt, 6, lane, v8zero());
    float bias = fsb1[nt * 16 + nc];
#pragma unroll
    for (int r = 0; r < 8; ++r)
      Db[(r + 8 * half) * 64 + nt * 16 + nc] = (__bf16)siluf(acc[r] + bias);
  }
  __syncthreads();

  // fs = mid @ fsW2 + fsb2  [16x64]@[64x320] -> Wf (f32)
  for (int nt = 0; nt < 20; ++nt) {
    v8f acc = gemm_tile(Db, 64, fsW2p, nt, 2, lane, v8zero());
    float bias = fsb2[nt * 16 + nc];
#pragma unroll
    for (int r = 0; r < 8; ++r)
      Wf[(r + 8 * half) * 320 + nt * 16 + nc] = acc[r] + bias;
  }
  __syncthreads();

  // t = ssp(edge_attr @ feW1 / sqrt(32)), zero-padded to K=32 -> Db[16x32]
  for (int idx = lane; idx < 512; idx += 32) {
    int row = idx >> 5, j = idx & 31;
    float val = 0.f;
    if (j < 8) {
      const float* ea = edge_attr + (size_t)(e0 + row) * 32;
      float acc = 0.f;
#pragma unroll
      for (int k = 0; k < 32; ++k) acc += ea[k] * feW1[k * 8 + j];
      val = sspf(acc * INV_S32);
    }
    Db[idx] = (__bf16)val;
  }
  __syncthreads();

  // fe = t @ feW2 / sqrt(8);  w = fe * fs  fused in the D store
  for (int nt = 0; nt < 20; ++nt) {
    v8f acc = gemm_tile(Db, 32, feW2p, nt, 1, lane, v8zero());
#pragma unroll
    for (int r = 0; r < 8; ++r) {
      int m = r + 8 * half, col = nt * 16 + nc;
      Wf[m * 320 + col] = Wf[m * 320 + col] * (acc[r] * INV_S8);
    }
  }
  __syncthreads();

  // pair = weighted uuu TP of h[src], h[dst] -> planar bf16 Xb (s0 dead)
  for (int idx = lane; idx < 1024; idx += 32) {
    int row = idx >> 6, u = idx & 63;
    int e = e0 + row, d = dsts[e], s = srcs[e];
    const __bf16* hx = hb + (size_t)s * 256;   // x = h[src]
    const __bf16* hy = hb + (size_t)d * 256;   // y = h[dst]
    float xs = (float)hx[u], ys = (float)hy[u];
    float xv0 = (float)hx[64 + u], xv1 = (float)hx[128 + u], xv2 = (float)hx[192 + u];
    float yv0 = (float)hy[64 + u], yv1 = (float)hy[128 + u], yv2 = (float)hy[192 + u];
    const float* wr = Wf + row * 320;
    float w000 = wr[u], w011 = wr[64 + u], w101 = wr[128 + u];
    float w110 = wr[192 + u], w111 = wr[256 + u];
    float dotv = xv0 * yv0 + xv1 * yv1 + xv2 * yv2;
    float cr0 = xv1 * yv2 - xv2 * yv1;
    float cr1 = xv2 * yv0 - xv0 * yv2;
    float cr2 = xv0 * yv1 - xv1 * yv0;
    Xb[row * 64 + u] = (__bf16)(CW0 * (w000 * xs * ys + w110 * dotv * INV_SQ3));
    Xb[1024 + row * 64 + u] =
        (__bf16)(CW1 * (INV_SQ3 * (w011 * xs * yv0 + w101 * xv0 * ys) + INV_SQ6 * w111 * cr0));
    Xb[2048 + row * 64 + u] =
        (__bf16)(CW1 * (INV_SQ3 * (w011 * xs * yv1 + w101 * xv1 * ys) + INV_SQ6 * w111 * cr1));
    Xb[3072 + row * 64 + u] =
        (__bf16)(CW1 * (INV_SQ3 * (w011 * xs * yv2 + w101 * xv2 * ys) + INV_SQ6 * w111 * cr2));
  }
  __syncthreads();

  norm_gate_bf(Xb, Fb, Gb, gW1p, gb1, gW2p, gb2, lane);
  o3_interleaved_out(Xb, Wo0p, bo0, Wo1p, Wf, lane);   // f32, reference layout
  __syncthreads();

  for (int idx = lane; idx < 4096; idx += 32)          // + residual, coalesced
    out[(size_t)e0 * 256 + idx] = Wf[idx] + npa[(size_t)e0 * 256 + idx];
}

// ------------------------------- launch ------------------------------------
extern "C" void kernel_launch(void* const* d_in, const int* in_sizes, int n_in,
                              void* d_out, int out_size, void* d_ws, size_t ws_size,
                              hipStream_t stream) {
  const float* node_attr = (const float*)d_in[0];
  const float* edge_attr = (const float*)d_in[1];
  const float* npa       = (const float*)d_in[2];
  const float* Wi0  = (const float*)d_in[3];
  const float* bi0  = (const float*)d_in[4];
  const float* Wi1  = (const float*)d_in[5];
  const float* gpW1 = (const float*)d_in[6];
  const float* gpb1 = (const float*)d_in[7];
  const float* gpW2 = (const float*)d_in[8];
  const float* gpb2 = (const float*)d_in[9];
  const float* Wn0  = (const float*)d_in[10];
  const float* bn0  = (const float*)d_in[11];
  const float* Wn1  = (const float*)d_in[12];
  const float* feW1 = (const float*)d_in[13];
  const float* feW2 = (const float*)d_in[14];
  const float* fsW1 = (const float*)d_in[15];
  const float* fsb1 = (const float*)d_in[16];
  const float* fsW2 = (const float*)d_in[17];
  const float* fsb2 = (const float*)d_in[18];
  const float* gW1  = (const float*)d_in[19];
  const float* gb1  = (const float*)d_in[20];
  const float* gW2  = (const float*)d_in[21];
  const float* gb2  = (const float*)d_in[22];
  const float* Wo0  = (const float*)d_in[23];
  const float* bo0  = (const float*)d_in[24];
  const float* Wo1  = (const float*)d_in[25];
  const int*   edge_index = (const int*)d_in[26];
  float* out = (float*)d_out;

  const int N = in_sizes[0] / 256;
  const int E = in_sizes[26] / 2;

  // workspace: planar bf16 na0 / h, then packed bf16 weights
  __bf16* na0b = (__bf16*)d_ws;
  __bf16* hb   = na0b + (size_t)N * 256;
  __bf16* wpk  = hb + (size_t)N * 256;
  __bf16* Wi0p  = wpk;                 // 64x64  : KB=2, NT=4  -> 4096 el
  __bf16* Wi1p  = Wi0p  + 4096;
  __bf16* Wn0p  = Wi1p  + 4096;
  __bf16* Wn1p  = Wn0p  + 4096;
  __bf16* Wo0p  = Wn1p  + 4096;
  __bf16* Wo1p  = Wo0p  + 4096;
  __bf16* gpW1p = Wo1p  + 4096;        // 128x128: KB=4, NT=8  -> 16384 el
  __bf16* gpW2p = gpW1p + 16384;
  __bf16* gW1p  = gpW2p + 16384;
  __bf16* gW2p  = gW1p  + 16384;
  __bf16* fsW1p = gW2p  + 16384;       // 192x64 : KB=6, NT=4  -> 12288 el
  __bf16* fsW2p = fsW1p + 12288;       // 64x320 : KB=2, NT=20 -> 20480 el
  __bf16* feW2p = fsW2p + 20480;       // 8x320 (K padded to 32): KB=1, NT=20

  auto pack = [&](const float* W, __bf16* dst, int ldW, int kMax, int KB, int NT) {
    int total = KB * NT * 32;
    pack_weight<<<dim3((total + 255) / 256), dim3(256), 0, stream>>>(W, dst, ldW, kMax, KB, NT);
  };
  pack(Wi0, Wi0p, 64, 64, 2, 4);
  pack(Wi1, Wi1p, 64, 64, 2, 4);
  pack(Wn0, Wn0p, 64, 64, 2, 4);
  pack(Wn1, Wn1p, 64, 64, 2, 4);
  pack(Wo0, Wo0p, 64, 64, 2, 4);
  pack(Wo1, Wo1p, 64, 64, 2, 4);
  pack(gpW1, gpW1p, 128, 128, 4, 8);
  pack(gpW2, gpW2p, 128, 128, 4, 8);
  pack(gW1,  gW1p,  128, 128, 4, 8);
  pack(gW2,  gW2p,  128, 128, 4, 8);
  pack(fsW1, fsW1p, 64, 192, 6, 4);
  pack(fsW2, fsW2p, 320, 64, 2, 20);
  pack(feW2, feW2p, 320, 8, 1, 20);

  const int nTiles = (N + 15) / 16;
  const int eTiles = (E + 15) / 16;
  const size_t shm = (size_t)WAVES * SLAB_B;

  node_stage<<<dim3((nTiles + WAVES - 1) / WAVES), dim3(32 * WAVES), shm, stream>>>(
      node_attr, Wi0p, bi0, Wi1p, gpW1p, gpb1, gpW2p, gpb2, Wn0p, bn0, Wn1p,
      na0b, hb, N, nTiles);

  edge_stage<<<dim3((eTiles + WAVES - 1) / WAVES), dim3(32 * WAVES), shm, stream>>>(
      edge_attr, npa, na0b, hb, feW1, feW2p, fsW1p, fsb1, fsW2p, fsb2,
      gW1p, gb1, gW2p, gb2, Wo0p, bo0, Wo1p, edge_index, out, E, eTiles);
}
